// H2OLlamaAttention_91130616087021
// MI455X (gfx1250) — matrix operand order repaired and checked by reading, not compiled, not run on hardware
//
#include <hip/hip_runtime.h>
#include <hip/hip_bf16.h>
#include <stdint.h>

#define S_LEN    2048
#define D_MODEL  4096
#define N_HEADS  32
#define N_KVH    8
#define HEAD_DIM 128
#define N_REP    (N_HEADS / N_KVH)

typedef __attribute__((ext_vector_type(16))) __bf16 v16bf;
typedef __attribute__((ext_vector_type(8)))  float  v8f;

union FragBF16 { v16bf v; uint4 q[2]; };

__device__ __forceinline__ unsigned short f32_to_bf16_rte(float x) {
  uint32_t u = __float_as_uint(x);
  u += 0x7FFFu + ((u >> 16) & 1u);
  return (unsigned short)(u >> 16);
}

// A fragment (16x32 bf16, MxK): lane<16 holds row=lane, K {0..7,16..23};
// lane>=16 holds row=lane-16, K {8..15,24..31}. Two 16B chunks per lane.
__device__ __forceinline__ v16bf load_frag_a(const unsigned short* base, int ld, int lane) {
  const int row  = lane & 15;
  const int koff = (lane < 16) ? 0 : 8;
  FragBF16 f;
  const unsigned short* p = base + (size_t)row * ld + koff;
  f.q[0] = *(const uint4*)(p);
  f.q[1] = *(const uint4*)(p + 16);
  return f.v;
}

// B fragment (32x16 bf16, KxN) loaded from an [N][K] row-major source:
// lane<16 holds col=lane, K {0..7,16..23}; lane>=16 holds col=lane-16, K {8..15,24..31}.
__device__ __forceinline__ v16bf load_frag_b(const unsigned short* base, int ld, int lane) {
  const int n    = lane & 15;
  const int koff = (lane < 16) ? 0 : 8;
  FragBF16 f;
  const unsigned short* p = base + (size_t)n * ld + koff;
  f.q[0] = *(const uint4*)(p);
  f.q[1] = *(const uint4*)(p + 16);
  return f.v;
}

__device__ __forceinline__ v8f wmma_bf16(v16bf a, v16bf b, v8f c) {
  return __builtin_amdgcn_wmma_f32_16x16x32_bf16(false, a, false, b, (short)0, c, false, false);
}

// ---------------------------------------------------------------- conversions
__global__ void k_f32_to_bf16(const float* __restrict__ in,
                              unsigned short* __restrict__ out, int n) {
  int i = blockIdx.x * blockDim.x + threadIdx.x;
  if (i < n) out[i] = f32_to_bf16_rte(in[i]);
}

__global__ void k_zero_f32(float* __restrict__ p, int n) {
  int i = blockIdx.x * blockDim.x + threadIdx.x;
  if (i < n) p[i] = 0.f;
}

// ---------------------------------------------------------------- GEMM (NT)
// C[M][N] = A[M][K] * W[N][K]^T ; bf16 in, f32 out.
// Block: 8 waves, block tile 64x256, wave tile 32x64 (2x4 WMMA tiles).
__global__ __launch_bounds__(256) void k_gemm_bf16_nt(
    const unsigned short* __restrict__ A,
    const unsigned short* __restrict__ W,
    float* __restrict__ C, int M, int N, int K) {
  const int lane = threadIdx.x & 31;
  const int wave = threadIdx.x >> 5;
  const int m0 = blockIdx.x * 64  + (wave >> 2) * 32;
  const int n0 = blockIdx.y * 256 + (wave & 3) * 64;

  const v8f vzero = {0.f,0.f,0.f,0.f,0.f,0.f,0.f,0.f};
  v8f acc[2][4];
  for (int i = 0; i < 2; ++i)
    for (int j = 0; j < 4; ++j) acc[i][j] = vzero;

  for (int k = 0; k < K; k += 32) {
    v16bf a0 = load_frag_a(A + (size_t)m0 * K + k, K, lane);
    v16bf a1 = load_frag_a(A + (size_t)(m0 + 16) * K + k, K, lane);
#pragma unroll
    for (int j = 0; j < 4; ++j) {
      v16bf b = load_frag_b(W + (size_t)(n0 + j * 16) * K + k, K, lane);
      acc[0][j] = wmma_bf16(a0, b, acc[0][j]);
      acc[1][j] = wmma_bf16(a1, b, acc[1][j]);
    }
  }
  const int half = (lane < 16) ? 0 : 8;
  const int col  = lane & 15;
  for (int i = 0; i < 2; ++i)
    for (int j = 0; j < 4; ++j)
#pragma unroll
      for (int r = 0; r < 8; ++r)
        C[(size_t)(m0 + i * 16 + r + half) * N + (n0 + j * 16 + col)] = acc[i][j][r];
}

// ---------------------------------------------------------------- RoPE + pack
// X: [S][nheads*HD] f32  ->  Xh: [nheads][S][HD] bf16 with RoPE applied.
__global__ void k_rope_pack(const float* __restrict__ X,
                            const int* __restrict__ pos_ids,
                            unsigned short* __restrict__ Xh, int nheads) {
  int idx = blockIdx.x * blockDim.x + threadIdx.x;
  int total = S_LEN * nheads * (HEAD_DIM / 2);
  if (idx >= total) return;
  int d = idx & 63;
  int h = (idx >> 6) % nheads;
  int s = idx / (64 * nheads);
  float pos = (float)pos_ids[s];
  // inv_freq = theta^(-2d/HD) = exp2(-(2d/HD)*log2(theta)); log2(10000)=13.2877124
  float inv_freq = exp2f(-(float)(2 * d) * (13.287712379549449f / (float)HEAD_DIM));
  float ang = pos * inv_freq;
  float c = cosf(ang), sn = sinf(ang);
  size_t src = (size_t)s * nheads * HEAD_DIM + (size_t)h * HEAD_DIM + d;
  float x1 = X[src], x2 = X[src + HEAD_DIM / 2];
  size_t dst = ((size_t)h * S_LEN + s) * HEAD_DIM + d;
  Xh[dst]                = f32_to_bf16_rte(x1 * c - x2 * sn);
  Xh[dst + HEAD_DIM / 2] = f32_to_bf16_rte(x2 * c + x1 * sn);
}

// V: [S][KVH*HD] f32  ->  Vt: [KVH][HD][S] bf16  (transposed so the P@V
// B-fragment reads are key-contiguous).
__global__ void k_v_pack(const float* __restrict__ V, unsigned short* __restrict__ Vt) {
  int idx = blockIdx.x * blockDim.x + threadIdx.x;
  if (idx >= S_LEN * N_KVH * HEAD_DIM) return;
  int s  = idx % S_LEN;
  int d  = (idx / S_LEN) % HEAD_DIM;
  int kv = idx / (S_LEN * HEAD_DIM);
  Vt[((size_t)kv * HEAD_DIM + d) * S_LEN + s] =
      f32_to_bf16_rte(V[(size_t)s * (N_KVH * HEAD_DIM) + (size_t)kv * HEAD_DIM + d]);
}

// ---------------------------------------------------------------- attention
// One block = one head x 128 query rows (8 waves x 16-row q-tiles).
__global__ __launch_bounds__(256) void k_attn(
    const unsigned short* __restrict__ Qh,  // [H][S][HD]
    const unsigned short* __restrict__ Kh,  // [KVH][S][HD]
    const unsigned short* __restrict__ Vt,  // [KVH][HD][S]
    float* __restrict__ O,                  // [S][H*HD] f32
    float* __restrict__ hh) {               // [H][S] f32, pre-zeroed
  __shared__ float hh_lds[S_LEN];
  __shared__ __align__(16) unsigned short p_lds[8][16 * 32];

  const int lane = threadIdx.x & 31;
  const int wave = threadIdx.x >> 5;
  const int h    = blockIdx.x >> 4;
  const int qb   = blockIdx.x & 15;
  const int kv   = h / N_REP;
  const int q0   = qb * 128 + wave * 16;

  const unsigned short* Qb = Qh + ((size_t)h  * S_LEN) * HEAD_DIM;
  const unsigned short* Kb = Kh + ((size_t)kv * S_LEN) * HEAD_DIM;
  const unsigned short* Vb = Vt + ((size_t)kv * HEAD_DIM) * S_LEN;

  for (int i = threadIdx.x; i < S_LEN; i += 256) hh_lds[i] = 0.f;
  __syncthreads();

  v16bf qf[4];
#pragma unroll
  for (int f = 0; f < 4; ++f)
    qf[f] = load_frag_a(Qb + (size_t)q0 * HEAD_DIM + f * 32, HEAD_DIM, lane);

  // 1/sqrt(HD) * log2(e): run softmax in the exp2 domain.
  const float kscale = 0.08838834764831845f * 1.4426950408889634f;
  const int half  = (lane < 16) ? 0 : 8;
  const int col   = lane & 15;
  const int qlast = q0 + 15;
  const v8f vzero = {0.f,0.f,0.f,0.f,0.f,0.f,0.f,0.f};

  float m[8], l[8];
#pragma unroll
  for (int r = 0; r < 8; ++r) { m[r] = -INFINITY; l[r] = 0.f; }

  // ---- Pass 1: per-row max + denominator (online, stats only) ----
  for (int kt = 0; kt <= qlast; kt += 16) {
    v8f c = vzero;
#pragma unroll
    for (int f = 0; f < 4; ++f) {
      v16bf b = load_frag_b(Kb + (size_t)kt * HEAD_DIM + f * 32, HEAD_DIM, lane);
      c = wmma_bf16(qf[f], b, c);
    }
#pragma unroll
    for (int r = 0; r < 8; ++r) {
      int row = q0 + r + half;
      int key = kt + col;
      float sv = (key <= row) ? c[r] * kscale : -INFINITY;
      float mt = sv;
#pragma unroll
      for (int off = 1; off < 16; off <<= 1) mt = fmaxf(mt, __shfl_xor(mt, off, 32));
      float mnew = fmaxf(m[r], mt);
      float p  = (key <= row) ? exp2f(sv - mnew) : 0.f;
      float ps = p;
#pragma unroll
      for (int off = 1; off < 16; off <<= 1) ps += __shfl_xor(ps, off, 32);
      l[r] = l[r] * exp2f(m[r] - mnew) + ps;
      m[r] = mnew;
    }
  }
  float invl[8];
#pragma unroll
  for (int r = 0; r < 8; ++r) invl[r] = 1.0f / l[r];

  // ---- Pass 2: normalized P, hh column sums, P @ V ----
  v8f o[8];
#pragma unroll
  for (int dt = 0; dt < 8; ++dt) o[dt] = vzero;

  unsigned short* pw = &p_lds[wave][0];

  for (int kb = 0; kb <= qlast; kb += 32) {
#pragma unroll
    for (int jn = 0; jn < 2; ++jn) {
      int kt = kb + jn * 16;
      if (kt <= qlast) {
        v8f c = vzero;
#pragma unroll
        for (int f = 0; f < 4; ++f) {
          v16bf b = load_frag_b(Kb + (size_t)kt * HEAD_DIM + f * 32, HEAD_DIM, lane);
          c = wmma_bf16(qf[f], b, c);
        }
        float csum = 0.f;
#pragma unroll
        for (int r = 0; r < 8; ++r) {
          int row = q0 + r + half;
          int key = kt + col;
          float p = 0.f;
          if (key <= row) p = exp2f(c[r] * kscale - m[r]) * invl[r];
          csum += p;
          pw[(r + half) * 32 + jn * 16 + col] = f32_to_bf16_rte(p);
        }
        csum += __shfl_xor(csum, 16, 32);      // combine the two row halves
        if (lane < 16) atomicAdd(&hh_lds[kt + lane], csum);
      } else {
#pragma unroll
        for (int r = 0; r < 8; ++r)
          pw[(r + half) * 32 + jn * 16 + col] = 0;
      }
    }
    asm volatile("s_wait_dscnt 0" ::: "memory");   // P staged in LDS
    v16bf pa = load_frag_a(pw, 32, lane);          // C-layout -> A-layout
#pragma unroll
    for (int dt = 0; dt < 8; ++dt) {
      v16bf vb = load_frag_b(Vb + (size_t)(dt * 16) * S_LEN + kb, S_LEN, lane);
      o[dt] = wmma_bf16(pa, vb, o[dt]);
    }
  }

#pragma unroll
  for (int dt = 0; dt < 8; ++dt)
#pragma unroll
    for (int r = 0; r < 8; ++r)
      O[(size_t)(q0 + r + half) * (N_HEADS * HEAD_DIM) + (size_t)h * HEAD_DIM + dt * 16 + col] = o[dt][r];

  __syncthreads();
  for (int i = threadIdx.x; i < S_LEN; i += 256)
    atomicAdd(&hh[(size_t)h * S_LEN + i], hh_lds[i]);
}

// ---------------------------------------------------------------- launch
extern "C" void kernel_launch(void* const* d_in, const int* in_sizes, int n_in,
                              void* d_out, int out_size, void* d_ws, size_t ws_size,
                              hipStream_t stream) {
  const float* hidden = (const float*)d_in[0];
  const float* Wq     = (const float*)d_in[1];
  const float* Wk     = (const float*)d_in[2];
  const float* Wv     = (const float*)d_in[3];
  const float* Wo     = (const float*)d_in[4];
  const int*   pos    = (const int*)d_in[5];

  char* ws = (char*)d_ws;
  size_t off = 0;
  auto carve = [&](size_t bytes) -> char* {
    char* p = ws + off; off += (bytes + 255) & ~(size_t)255; return p;
  };

  const size_t QN = (size_t)N_HEADS * HEAD_DIM;  // 4096
  const size_t KN = (size_t)N_KVH  * HEAD_DIM;   // 1024

  unsigned short* h_bf  = (unsigned short*)carve((size_t)S_LEN * D_MODEL * 2);
  unsigned short* wq_bf = (unsigned short*)carve(QN * D_MODEL * 2);
  unsigned short* wk_bf = (unsigned short*)carve(KN * D_MODEL * 2);
  unsigned short* wv_bf = (unsigned short*)carve(KN * D_MODEL * 2);
  unsigned short* wo_bf = (unsigned short*)carve((size_t)D_MODEL * QN * 2);
  float*          qf32  = (float*)carve((size_t)S_LEN * QN * 4);
  float*          kf32  = (float*)carve((size_t)S_LEN * KN * 4);
  float*          vf32  = (float*)carve((size_t)S_LEN * KN * 4);
  unsigned short* qh    = (unsigned short*)carve((size_t)N_HEADS * S_LEN * HEAD_DIM * 2);
  unsigned short* kh    = (unsigned short*)carve((size_t)N_KVH  * S_LEN * HEAD_DIM * 2);
  unsigned short* vt    = (unsigned short*)carve((size_t)N_KVH  * HEAD_DIM * S_LEN * 2);
  float*          ao32  = (float*)carve((size_t)S_LEN * QN * 4);
  unsigned short* ao_bf = (unsigned short*)carve((size_t)S_LEN * QN * 2);

  auto cvt = [&](const float* src, unsigned short* dst, size_t n) {
    k_f32_to_bf16<<<(unsigned)((n + 255) / 256), 256, 0, stream>>>(src, dst, (int)n);
  };
  cvt(hidden, h_bf, (size_t)S_LEN * D_MODEL);
  cvt(Wq, wq_bf, QN * D_MODEL);
  cvt(Wk, wk_bf, KN * D_MODEL);
  cvt(Wv, wv_bf, KN * D_MODEL);
  cvt(Wo, wo_bf, (size_t)D_MODEL * QN);

  // Projections: C = hidden * W^T
  k_gemm_bf16_nt<<<dim3(S_LEN / 64, (unsigned)(QN / 256)), 256, 0, stream>>>(
      h_bf, wq_bf, qf32, S_LEN, (int)QN, D_MODEL);
  k_gemm_bf16_nt<<<dim3(S_LEN / 64, (unsigned)(KN / 256)), 256, 0, stream>>>(
      h_bf, wk_bf, kf32, S_LEN, (int)KN, D_MODEL);
  k_gemm_bf16_nt<<<dim3(S_LEN / 64, (unsigned)(KN / 256)), 256, 0, stream>>>(
      h_bf, wv_bf, vf32, S_LEN, (int)KN, D_MODEL);

  // RoPE + head-major bf16 packing; V transposed to [KVH][HD][S].
  k_rope_pack<<<(S_LEN * N_HEADS * 64 + 255) / 256, 256, 0, stream>>>(qf32, pos, qh, N_HEADS);
  k_rope_pack<<<(S_LEN * N_KVH  * 64 + 255) / 256, 256, 0, stream>>>(kf32, pos, kh, N_KVH);
  k_v_pack<<<(S_LEN * N_KVH * HEAD_DIM + 255) / 256, 256, 0, stream>>>(vf32, vt);

  float* out = (float*)d_out;
  float* hh  = out + (size_t)S_LEN * QN;   // second tuple element
  k_zero_f32<<<(N_HEADS * S_LEN + 255) / 256, 256, 0, stream>>>(hh, N_HEADS * S_LEN);

  k_attn<<<N_HEADS * (S_LEN / 128), 256, 0, stream>>>(qh, kh, vt, ao32, hh);

  // Output projection: out = attn_out * Wo^T
  cvt(ao32, ao_bf, (size_t)S_LEN * QN);
  k_gemm_bf16_nt<<<dim3(S_LEN / 64, D_MODEL / 256), 256, 0, stream>>>(
      ao_bf, wo_bf, out, S_LEN, D_MODEL, (int)QN);
}
